// FrequencyAttention_23673859735910
// MI455X (gfx1250) — matrix operand order repaired and checked by reading, not compiled, not run on hardware
//
#include <hip/hip_runtime.h>
#include <hip/hip_bf16.h>

typedef __attribute__((ext_vector_type(16))) _Float16 v16h;
typedef __attribute__((ext_vector_type(8)))  _Float16 v8h;
typedef __attribute__((ext_vector_type(8)))  float    v8f;
typedef unsigned int v4u __attribute__((ext_vector_type(4)));
typedef int          v8i __attribute__((ext_vector_type(8)));
typedef int          v4i __attribute__((ext_vector_type(4)));

#define PI2 6.28318530717958647692f

#if __has_builtin(__builtin_amdgcn_tensor_load_to_lds) && \
    __has_builtin(__builtin_amdgcn_s_wait_tensorcnt)
#define USE_TDM 1
#else
#define USE_TDM 0
#endif

// ---------------------------------------------------------------------------
// 1) Repack conv2 weights (D=128, C=256, 3, 3) f32 -> f16 [kykx][d][c]
// ---------------------------------------------------------------------------
__global__ __launch_bounds__(256) void k_pack_w(const float* __restrict__ w,
                                                _Float16* __restrict__ wH) {
  int idx = blockIdx.x * 256 + threadIdx.x;   // 9*128*256 = 294912 exactly
  int kk = idx >> 15;                         // tap index 0..8
  int r  = idx & 32767;
  int d  = r >> 8;
  int c  = r & 255;
  int ky = kk / 3, kx = kk - ky * 3;
  wH[idx] = (_Float16)w[(((d * 256 + c) * 3 + ky) * 3 + kx)];
}

// ---------------------------------------------------------------------------
// 2) Dilated 3x3 conv (dil=3, pad=3) + BN + ReLU as implicit GEMM via WMMA.
//    Block = (b, y): M=64 x-positions, N=128 channels, K=2304.
//    8 waves: 4 (M) x 2 (N); each wave: 16M x 64N = 4 accumulators.
//    B tile staged by the Tensor Data Mover (hardware LDS row padding),
//    A tile staged by all threads (needs zero-fill for image borders).
// ---------------------------------------------------------------------------
__global__ __launch_bounds__(256) void k_conv_wmma(
    const float* __restrict__ x, const _Float16* __restrict__ wH,
    const float* __restrict__ conv2_b, const float* __restrict__ bn_sc,
    const float* __restrict__ bn_bi, float* __restrict__ out) {
  __shared__ __align__(16) _Float16 ldsA[64 * 40];    // [pos][c], pad to 40
  __shared__ __align__(16) _Float16 ldsB[128 * 40];   // [d][c],   pad to 40
  __shared__ float ldsC[128 * 64];                    // [d][pos] store bounce
  const int tid  = threadIdx.x;
  const int b    = blockIdx.x >> 6;
  const int y    = blockIdx.x & 63;
  const int wave = tid >> 5, lane = tid & 31;
  const int wm = wave & 3, wn = wave >> 2;
  const int hi = lane >> 4, ln = lane & 15;

  v8f acc0 = {}, acc1 = {}, acc2 = {}, acc3 = {};

  for (int kk = 0; kk < 9; ++kk) {
    const int ky = kk / 3, kx = kk - ky * 3;
    const int ys = y + 3 * (ky - 1);
    const int dx = 3 * (kx - 1);
    const bool rowok = ((unsigned)ys < 64u);
    for (int cb = 0; cb < 256; cb += 32) {
      __syncthreads();    // previous iteration's LDS reads complete
      // ---- Stage A tile: 64 pos x 32 c (zero-pad out-of-image taps) ----
#pragma unroll
      for (int r = 0; r < 8; ++r) {
        int idx = tid + 256 * r;            // 0..2047
        int cc  = idx >> 6, pos = idx & 63;
        int xs  = pos + dx;
        float v = 0.f;
        if (rowok && (unsigned)xs < 64u)
          v = x[(((size_t)b * 256 + (cb + cc)) * 64 + ys) * 64 + xs];
        ldsA[pos * 40 + cc] = (_Float16)v;
      }
      // ---- Stage B tile: 128 d x 32 c ----
#if USE_TDM
      if (wave == 0) {
        // 2D tile: 32 halves/row (16 DW), 128 rows, row stride 256 halves.
        // LDS padding: +4 DW after every 16 DW -> 40-half padded rows.
        unsigned long long gaddr =
            (unsigned long long)(size_t)(wH + (size_t)kk * 32768 + cb);
        unsigned ldsb = (unsigned)(size_t)(void*)&ldsB[0];
        v4u g0;
        g0[0] = 1u;                               // count=1 valid descriptor
        g0[1] = ldsb;                             // lds_addr
        g0[2] = (unsigned)(gaddr & 0xffffffffu);  // global_addr[31:0]
        g0[3] = (unsigned)((gaddr >> 32) & 0x01ffffffu) | (2u << 30); // type=2
        v8i g1;
        g1[0] = (1 << 16)      // data_size = 2 bytes
              | (1 << 20)      // pad_enable
              | (3 << 22)      // pad_interval: 16 DWORDs
              | (3 << 25);     // pad_amount:   4 DWORDs
        g1[1] = (256 << 16);   // tensor_dim0 = 256 (low 16 in bits 63:48)
        g1[2] = (128 << 16);   // tensor_dim0 hi=0 | tensor_dim1 = 128
        g1[3] = (32 << 16);    // tensor_dim1 hi=0 | tile_dim0 = 32
        g1[4] = 128;           // tile_dim1 = 128 | tile_dim2 = 0
        g1[5] = 256;           // tensor_dim0_stride = 256
        g1[6] = 0;
        g1[7] = 0;
        v4i z4 = {0, 0, 0, 0};               // groups 2/3 unused (2D tensor)
        v8i z8 = {0, 0, 0, 0, 0, 0, 0, 0};   // trailing group (clang-23 form)
        __builtin_amdgcn_tensor_load_to_lds(g0, g1, z4, z4, z8, 0);
        __builtin_amdgcn_s_wait_tensorcnt(0);
      }
#else
#pragma unroll
      for (int r = 0; r < 16; ++r) {
        int idx = tid + 256 * r;            // 0..4095
        int d = idx >> 5, cc = idx & 31;
        ldsB[d * 40 + cc] = wH[((size_t)kk * 128 + d) * 256 + cb + cc];
      }
#endif
      __syncthreads();
      // A fragment: lanes<16: K {0..7,16..23}; lanes>=16: K {8..15,24..31}
      const _Float16* ap = &ldsA[(wm * 16 + ln) * 40 + 8 * hi];
      v8h alo = *(const v8h*)(ap);
      v8h ahi = *(const v8h*)(ap + 16);
      v16h a;
#pragma unroll
      for (int e = 0; e < 8; ++e) { a[e] = alo[e]; a[e + 8] = ahi[e]; }
#pragma unroll
      for (int nt = 0; nt < 4; ++nt) {
        // B fragment: lanes<16: K {0..7,16..23}; lanes>=16: K {8..15,24..31}
        const _Float16* bp = &ldsB[(wn * 64 + nt * 16 + ln) * 40 + 8 * hi];
        v8h blo = *(const v8h*)(bp);
        v8h bhi = *(const v8h*)(bp + 16);
        v16h bf;
#pragma unroll
        for (int e = 0; e < 8; ++e) { bf[e] = blo[e]; bf[e + 8] = bhi[e]; }
        v8f& acc = (nt == 0) ? acc0 : (nt == 1) ? acc1 : (nt == 2) ? acc2 : acc3;
        acc = __builtin_amdgcn_wmma_f32_16x16x32_f16(
            false, a, false, bf, (short)0, acc, false, false);
      }
    }
  }
  // Epilogue: (conv + b)*scale + bias, ReLU; bounce through LDS so global
  // stores are contiguous 64-float rows. C/D layout: VGPR r -> M=r+8*hi.
#pragma unroll
  for (int nt = 0; nt < 4; ++nt) {
    const v8f acc = (nt == 0) ? acc0 : (nt == 1) ? acc1 : (nt == 2) ? acc2 : acc3;
    int d = wn * 64 + nt * 16 + ln;
    float sc = bn_sc[d];
    float bi = conv2_b[d] * sc + bn_bi[d];
#pragma unroll
    for (int r = 0; r < 8; ++r) {
      int pos = wm * 16 + r + 8 * hi;
      float v = acc[r] * sc + bi;
      ldsC[d * 64 + pos] = v > 0.f ? v : 0.f;
    }
  }
  __syncthreads();
  const size_t obase = (((size_t)b * 128) * 64 + y) * 64;
#pragma unroll
  for (int r = 0; r < 32; ++r) {
    int idx = tid + 256 * r;                // 0..8191 ; idx = d*64 + pos
    int d = idx >> 6;
    out[obase + (size_t)d * 4096 + (idx & 63)] = ldsC[idx];
  }
}

// ---------------------------------------------------------------------------
// LDS FFT helpers (radix-2 DIT, 256 threads cooperating)
// ---------------------------------------------------------------------------
__device__ __forceinline__ void fft64_pass(float* re, float* im, int rs, int js,
                                           float sign, int tid) {
#pragma unroll
  for (int it = 0; it < 16; ++it) {
    int idx = tid + 256 * it;
    int row = idx >> 6, j = idx & 63;
    int jr = (int)(__brev((unsigned)j) >> 26);
    if (j < jr) {
      int a = row * rs + j * js, c = row * rs + jr * js;
      float tr = re[a]; re[a] = re[c]; re[c] = tr;
      float ti = im[a]; im[a] = im[c]; im[c] = ti;
    }
  }
  __syncthreads();
  for (int s = 1; s <= 6; ++s) {
    const int m = 1 << s, half = m >> 1;
#pragma unroll
    for (int it = 0; it < 8; ++it) {
      int idx = tid + 256 * it;             // 2048 butterflies/stage
      int row = idx >> 5, k = idx & 31;
      int grp = k >> (s - 1);
      int j   = k & (half - 1);
      int i1  = row * rs + (grp * m + j) * js;
      int i2  = i1 + half * js;
      float sn, cs;
      __sincosf(sign * PI2 * (float)j / (float)m, &sn, &cs);
      float xr = re[i2], xi = im[i2];
      float tr = xr * cs - xi * sn, ti = xr * sn + xi * cs;
      float ur = re[i1], ui = im[i1];
      re[i1] = ur + tr; im[i1] = ui + ti;
      re[i2] = ur - tr; im[i2] = ui - ti;
    }
    __syncthreads();
  }
}

__device__ __forceinline__ void ifft4096(float* re, float* im, int tid) {
#pragma unroll
  for (int it = 0; it < 16; ++it) {
    int j  = tid + 256 * it;
    int jr = (int)(__brev((unsigned)j) >> 20);
    if (j < jr) {
      float tr = re[j]; re[j] = re[jr]; re[jr] = tr;
      float ti = im[j]; im[j] = im[jr]; im[jr] = ti;
    }
  }
  __syncthreads();
  for (int s = 1; s <= 12; ++s) {
    const int m = 1 << s, half = m >> 1;
#pragma unroll
    for (int it = 0; it < 8; ++it) {
      int k   = tid + 256 * it;             // 2048 butterflies/stage
      int grp = k >> (s - 1);
      int j   = k & (half - 1);
      int i1  = grp * m + j, i2 = i1 + half;
      float sn, cs;
      __sincosf(PI2 * (float)j / (float)m, &sn, &cs);
      float xr = re[i2], xi = im[i2];
      float tr = xr * cs - xi * sn, ti = xr * sn + xi * cs;
      float ur = re[i1], ui = im[i1];
      re[i1] = ur + tr; im[i1] = ui + ti;
      re[i2] = ur - tr; im[i2] = ui - ti;
    }
    __syncthreads();
  }
}

// ---------------------------------------------------------------------------
// 3) Forward 2D FFT of conv2, one (b,c) 64x64 image per block.
// ---------------------------------------------------------------------------
__global__ __launch_bounds__(256) void k_fft2_fwd(const float* __restrict__ src,
                                                  float* __restrict__ fRe,
                                                  float* __restrict__ fIm) {
  __shared__ float re[4096];
  __shared__ float im[4096];
  const int tid = threadIdx.x;
  const size_t base = (size_t)blockIdx.x * 4096;
#pragma unroll
  for (int it = 0; it < 16; ++it) {
    int idx = tid + 256 * it;
    re[idx] = src[base + idx];
    im[idx] = 0.f;
  }
  __syncthreads();
  fft64_pass(re, im, 64, 1, -1.f, tid);   // along w
  fft64_pass(re, im, 1, 64, -1.f, tid);   // along h
#pragma unroll
  for (int it = 0; it < 16; ++it) {
    int idx = tid + 256 * it;
    fRe[base + idx] = re[idx];
    fIm[base + idx] = im[idx];
  }
}

// ---------------------------------------------------------------------------
// 4) Per (b,head): 16x16 complex Gram (no conjugate, per einsum) + row norms^2
// ---------------------------------------------------------------------------
__global__ __launch_bounds__(256) void k_gram(const float* __restrict__ fRe,
                                              const float* __restrict__ fIm,
                                              float* __restrict__ GR,
                                              float* __restrict__ GI,
                                              float* __restrict__ N2) {
  __shared__ float sr[16 * 128];
  __shared__ float si[16 * 128];
  const int tid = threadIdx.x;
  const int bh  = blockIdx.x;                     // b*8 + h
  const size_t fbase = (size_t)bh * 16 * 4096;    // == (b*128 + h*16)*4096
  const int i = tid >> 4, j = tid & 15;
  float aR = 0.f, aI = 0.f, n2 = 0.f;
  for (int cb = 0; cb < 4096; cb += 128) {
    __syncthreads();
#pragma unroll
    for (int r = 0; r < 8; ++r) {
      int idx = tid + 256 * r;                    // 2048
      int row = idx >> 7, n = idx & 127;
      sr[row * 128 + n] = fRe[fbase + (size_t)row * 4096 + cb + n];
      si[row * 128 + n] = fIm[fbase + (size_t)row * 4096 + cb + n];
    }
    __syncthreads();
    for (int n = 0; n < 128; ++n) {
      float ri = sr[i * 128 + n], ii = si[i * 128 + n];
      float rj = sr[j * 128 + n], ij = si[j * 128 + n];
      aR += ri * rj - ii * ij;
      aI += ri * ij + ii * rj;
    }
    if (tid < 16) {
      for (int n = 0; n < 128; ++n) {
        float r0 = sr[tid * 128 + n], i0 = si[tid * 128 + n];
        n2 += r0 * r0 + i0 * i0;
      }
    }
  }
  GR[bh * 256 + tid] = aR;
  GI[bh * 256 + tid] = aI;
  if (tid < 16) N2[bh * 16 + tid] = n2;
}

// ---------------------------------------------------------------------------
// 5) Normalize, temperature, dual softmax, fold (1/16)*IDFT16 -> M matrices
// ---------------------------------------------------------------------------
__global__ __launch_bounds__(256) void k_attn_fold(
    const float* __restrict__ GR, const float* __restrict__ GI,
    const float* __restrict__ N2, const float* __restrict__ temp,
    float* __restrict__ MR, float* __restrict__ MI) {
  __shared__ float aR[256], aI[256];
  __shared__ float mxR[16], smR[16], mxI[16], smI[16];
  const int tid = threadIdx.x;
  const int bh  = blockIdx.x;
  const int h   = bh & 7;
  const int i   = tid >> 4, j = tid & 15;
  float ni = fmaxf(sqrtf(N2[bh * 16 + i]), 1e-12f);
  float nj = fmaxf(sqrtf(N2[bh * 16 + j]), 1e-12f);
  float inv = temp[h] / (ni * nj);
  aR[tid] = GR[bh * 256 + tid] * inv;
  aI[tid] = GI[bh * 256 + tid] * inv;
  __syncthreads();
  if (j == 0) {
    float m = -3.4e38f;
    for (int q = 0; q < 16; ++q) m = fmaxf(m, aR[i * 16 + q]);
    float s = 0.f;
    for (int q = 0; q < 16; ++q) s += expf(aR[i * 16 + q] - m);
    mxR[i] = m; smR[i] = s;
  } else if (j == 1) {
    float m = -3.4e38f;
    for (int q = 0; q < 16; ++q) m = fmaxf(m, aI[i * 16 + q]);
    float s = 0.f;
    for (int q = 0; q < 16; ++q) s += expf(aI[i * 16 + q] - m);
    mxI[i] = m; smI[i] = s;
  }
  __syncthreads();
  float pr = expf(aR[tid] - mxR[i]) / smR[i];
  float pi = expf(aI[tid] - mxI[i]) / smI[i];
  aR[tid] = pr; aI[tid] = pi;                 // own slot only: no race
  __syncthreads();
  // M[i][j] = (1/16) sum_k exp(+i*2pi*i*k/16) * attn[k][j]
  float mr = 0.f, mi = 0.f;
  for (int k = 0; k < 16; ++k) {
    float sn, cs;
    __sincosf(PI2 * (float)(i * k) * (1.f / 16.f), &sn, &cs);
    float ar = aR[k * 16 + j], ai = aI[k * 16 + j];
    mr += cs * ar - sn * ai;
    mi += cs * ai + sn * ar;
  }
  MR[bh * 256 + tid] = mr * (1.f / 16.f);
  MI[bh * 256 + tid] = mi * (1.f / 16.f);
}

// ---------------------------------------------------------------------------
// 6) Gating MLP on f.real: 128 -> 8 (BN+ReLU) -> 128 -> sigmoid, per (b,u,v)
// ---------------------------------------------------------------------------
__global__ __launch_bounds__(256) void k_gate(
    const float* __restrict__ fRe, const float* __restrict__ w1,
    const float* __restrict__ w1b, const float* __restrict__ bnsc,
    const float* __restrict__ bnbi, const float* __restrict__ w2,
    const float* __restrict__ w2b, float* __restrict__ g) {
  __shared__ float W1[1024], W2[1024];
  const int tid = threadIdx.x;
#pragma unroll
  for (int r = 0; r < 4; ++r) {
    W1[tid + 256 * r] = w1[tid + 256 * r];
    W2[tid + 256 * r] = w2[tid + 256 * r];
  }
  __syncthreads();
  const int gidx = blockIdx.x * 256 + tid;    // 0..65535
  const int b = gidx >> 12, p = gidx & 4095;
  float hs[8];
#pragma unroll
  for (int q = 0; q < 8; ++q) hs[q] = w1b[q];
  for (int c = 0; c < 128; ++c) {
    float fr = fRe[(((size_t)b * 128 + c) << 12) + p];
#pragma unroll
    for (int q = 0; q < 8; ++q) hs[q] += W1[q * 128 + c] * fr;
  }
#pragma unroll
  for (int q = 0; q < 8; ++q) {
    float v = hs[q] * bnsc[q] + bnbi[q];
    hs[q] = v > 0.f ? v : 0.f;
  }
  for (int c = 0; c < 128; ++c) {
    float o = w2b[c];
#pragma unroll
    for (int q = 0; q < 8; ++q) o += W2[c * 8 + q] * hs[q];
    g[(((size_t)b * 128 + c) << 12) + p] = 1.f / (1.f + expf(-o));
  }
}

// ---------------------------------------------------------------------------
// 7) out_f branch: row i of (M @ qkv) -> 4096-pt IFFT -> |.| -> out[:,0:128]+x
// ---------------------------------------------------------------------------
__global__ __launch_bounds__(256) void k_attn_apply(
    const float* __restrict__ fRe, const float* __restrict__ fIm,
    const float* __restrict__ MR, const float* __restrict__ MI,
    const float* __restrict__ x, float* __restrict__ out) {
  __shared__ float re[4096], im[4096];
  __shared__ float mr[16], mi[16];
  const int tid = threadIdx.x;
  const int gid = blockIdx.x;                 // bh*16 + i
  const int i = gid & 15, bh = gid >> 4;
  const size_t fbase = (size_t)bh * 16 * 4096;
  if (tid < 16) {
    mr[tid] = MR[bh * 256 + i * 16 + tid];
    mi[tid] = MI[bh * 256 + i * 16 + tid];
  }
  __syncthreads();
  for (int it = 0; it < 16; ++it) {
    int n = tid + 256 * it;
    float ar = 0.f, ai = 0.f;
#pragma unroll
    for (int jj = 0; jj < 16; ++jj) {
      float fr = fRe[fbase + (size_t)jj * 4096 + n];
      float fi = fIm[fbase + (size_t)jj * 4096 + n];
      ar += mr[jj] * fr - mi[jj] * fi;
      ai += mr[jj] * fi + mi[jj] * fr;
    }
    re[n] = ar; im[n] = ai;
  }
  __syncthreads();
  ifft4096(re, im, tid);
  const int b = bh >> 3;
  const int cc = (bh & 7) * 16 + i;           // channel in [0,128)
  const size_t obase = ((size_t)b * 256 + cc) * 4096;
#pragma unroll
  for (int it = 0; it < 16; ++it) {
    int n = tid + 256 * it;
    float v = sqrtf(re[n] * re[n] + im[n] * im[n]) * (1.f / 4096.f);
    out[obase + n] = v + x[obase + n];
  }
}

// ---------------------------------------------------------------------------
// 8) out_l branch: g*f -> 2D IFFT(64x64) -> |.| -> out[:,128:256] + x
// ---------------------------------------------------------------------------
__global__ __launch_bounds__(256) void k_outl(
    const float* __restrict__ fRe, const float* __restrict__ fIm,
    const float* __restrict__ g, const float* __restrict__ x,
    float* __restrict__ out) {
  __shared__ float re[4096], im[4096];
  const int tid = threadIdx.x;
  const size_t base = (size_t)blockIdx.x * 4096;    // (b*128 + c)
#pragma unroll
  for (int it = 0; it < 16; ++it) {
    int n = tid + 256 * it;
    float gg = g[base + n];
    re[n] = gg * fRe[base + n];
    im[n] = gg * fIm[base + n];
  }
  __syncthreads();
  fft64_pass(re, im, 64, 1, 1.f, tid);
  fft64_pass(re, im, 1, 64, 1.f, tid);
  const int b = blockIdx.x >> 7, c = blockIdx.x & 127;
  const size_t obase = ((size_t)b * 256 + 128 + c) * 4096;
#pragma unroll
  for (int it = 0; it < 16; ++it) {
    int n = tid + 256 * it;
    float v = sqrtf(re[n] * re[n] + im[n] * im[n]) * (1.f / 4096.f);
    out[obase + n] = v + x[obase + n];
  }
}

// ---------------------------------------------------------------------------
extern "C" void kernel_launch(void* const* d_in, const int* in_sizes, int n_in,
                              void* d_out, int out_size, void* d_ws,
                              size_t ws_size, hipStream_t stream) {
  const float* x       = (const float*)d_in[0];
  const float* conv2_w = (const float*)d_in[1];
  const float* conv2_b = (const float*)d_in[2];
  const float* bn2_sc  = (const float*)d_in[3];
  const float* bn2_bi  = (const float*)d_in[4];
  const float* temp    = (const float*)d_in[5];
  const float* w1      = (const float*)d_in[6];
  const float* w1b     = (const float*)d_in[7];
  const float* bnwsc   = (const float*)d_in[8];
  const float* bnwbi   = (const float*)d_in[9];
  const float* w2      = (const float*)d_in[10];
  const float* w2b     = (const float*)d_in[11];
  float* out = (float*)d_out;

  char* ws = (char*)d_ws;
  const size_t MB32 = 33554432u;              // 8M floats
  _Float16* wH      = (_Float16*)ws;          // 576 KB used, 1 MB reserved
  float* conv2buf   = (float*)(ws + (1u << 20));            // reused as g
  float* fRe        = (float*)(ws + (1u << 20) + MB32);
  float* fIm        = (float*)(ws + (1u << 20) + 2 * MB32);
  char*  tail       = ws + (1u << 20) + 3 * MB32;
  float* GR = (float*)(tail);
  float* GI = (float*)(tail + 131072);
  float* N2 = (float*)(tail + 262144);
  float* MR = (float*)(tail + 262144 + 8192);
  float* MI = (float*)(tail + 262144 + 8192 + 131072);

  k_pack_w   <<<1152, 256, 0, stream>>>(conv2_w, wH);
  k_conv_wmma<<<1024, 256, 0, stream>>>(x, wH, conv2_b, bn2_sc, bn2_bi, conv2buf);
  k_fft2_fwd <<<2048, 256, 0, stream>>>(conv2buf, fRe, fIm);
  k_gram     <<<128,  256, 0, stream>>>(fRe, fIm, GR, GI, N2);
  k_attn_fold<<<128,  256, 0, stream>>>(GR, GI, N2, temp, MR, MI);
  k_gate     <<<256,  256, 0, stream>>>(fRe, w1, w1b, bnwsc, bnwbi, w2, w2b,
                                        conv2buf);   // conv2buf recycled as g
  k_attn_apply<<<2048, 256, 0, stream>>>(fRe, fIm, MR, MI, x, out);
  k_outl     <<<2048, 256, 0, stream>>>(fRe, fIm, conv2buf, x, out);
  (void)in_sizes; (void)n_in; (void)out_size; (void)ws_size;
}
